// MultiModalSDTPS_48859547959316
// MI455X (gfx1250) — matrix operand order repaired and checked by reading, not compiled, not run on hardware
//
#include <hip/hip_runtime.h>
#include <hip/hip_bf16.h>
#include <math.h>

// Sizes from the reference
#define B_ 128
#define N_ 196
#define C_ 768
#define H_ 4
#define D_ 192
#define KEEP_ 118   // ceil(196*0.6)

typedef __attribute__((ext_vector_type(16))) __bf16 v16bf;
typedef __attribute__((ext_vector_type(8)))  float  v8f;

// module -> global index: order = [(0,0),(0,1),(0,2),(1,1),(1,0),(1,2),(2,2),(2,0),(2,1)]
__device__ __constant__ int c_gi[9] = {0,1,2, 1,0,2, 2,0,1};

// ---------------------------------------------------------------------------
// Kernel A1: Q[i] (128x768) = G[gi[i]] (128x768) @ qw[i]^T + qb[i]
// One wave per 16x32 output tile (two 16x16 WMMA tiles sharing the A frag).
// bf16 WMMA, f32 accumulate.
// ---------------------------------------------------------------------------
__global__ __launch_bounds__(32) void k_qproj(
    const float* __restrict__ g0, const float* __restrict__ g1, const float* __restrict__ g2,
    const float* __restrict__ qw, const float* __restrict__ qb,
    float* __restrict__ Q)
{
  const int tile = blockIdx.x;            // i*(8*24) + mt*24 + ntp
  const int i   = tile / (8*24);
  const int r   = tile % (8*24);
  const int mt  = r / 24, ntp = r % 24;   // ntp selects a 32-wide column pair
  const int lane  = threadIdx.x;
  const int half  = lane >> 4;
  const int lan15 = lane & 15;
  const float* G = (c_gi[i]==0) ? g0 : (c_gi[i]==1) ? g1 : g2;
  const int m  = mt*16 + lan15;
  const int n0 = ntp*32 + lan15;          // first 16-col tile
  const int n1 = n0 + 16;                 // second 16-col tile
  const int koffA = half ? 16 : 0;        // A: lanes16-31 hold K 16..31
  const int koffB = half ? 16 : 0;        // B: lanes16-31 hold K 16..31
  const float* Arow  = G  + (size_t)m*C_;
  const float* Brow0 = qw + ((size_t)i*C_ + n0)*C_;  // row n of qw == column n of qw^T
  const float* Brow1 = qw + ((size_t)i*C_ + n1)*C_;
  v8f acc0 = {}, acc1 = {};
  #pragma unroll 4
  for (int k0 = 0; k0 < C_; k0 += 32) {
    v16bf a, b0, b1;
    const float* ap = Arow + k0 + koffA;
    #pragma unroll
    for (int e = 0; e < 8; ++e) { a[e] = (__bf16)ap[e]; a[8+e] = (__bf16)ap[8+e]; }
    const float* bp0 = Brow0 + k0 + koffB;
    const float* bp1 = Brow1 + k0 + koffB;
    #pragma unroll
    for (int e = 0; e < 16; ++e) { b0[e] = (__bf16)bp0[e]; b1[e] = (__bf16)bp1[e]; }
    acc0 = __builtin_amdgcn_wmma_f32_16x16x32_bf16(false, a, false, b0, (short)0, acc0, false, false);
    acc1 = __builtin_amdgcn_wmma_f32_16x16x32_bf16(false, a, false, b1, (short)0, acc1, false, false);
  }
  const float bias0 = qb[i*C_ + n0];
  const float bias1 = qb[i*C_ + n1];
  #pragma unroll
  for (int rr = 0; rr < 8; ++rr) {
    int row = mt*16 + half*8 + rr;        // D layout: VGPR rr -> M = rr (+8 for upper half)
    Q[((size_t)i*B_ + row)*C_ + n0] = acc0[rr] + bias0;
    Q[((size_t)i*B_ + row)*C_ + n1] = acc1[rr] + bias1;
  }
}

// ---------------------------------------------------------------------------
// Kernel A2: W[i][b][h][:] = Q[i][b, h*192 : (h+1)*192] @ kw[i][h*192: , :]
// 36 GEMMs of (128x192)@(192x768); one wave per 16x32 tile; bf16 WMMA.
// ---------------------------------------------------------------------------
__global__ __launch_bounds__(32) void k_wproj(
    const float* __restrict__ Q, const float* __restrict__ kw,
    float* __restrict__ W)
{
  const int tile = blockIdx.x;            // (i*4+h)*(8*24) + mt*24 + ntp
  const int ih = tile / (8*24);
  const int i  = ih >> 2, h = ih & 3;
  const int r  = tile % (8*24);
  const int mt = r / 24, ntp = r % 24;
  const int lane  = threadIdx.x;
  const int half  = lane >> 4;
  const int lan15 = lane & 15;
  const int m  = mt*16 + lan15;
  const int n0 = ntp*32 + lan15;
  const int koffA = half ? 16 : 0;
  const int koffB = half ? 16 : 0;
  const float* Arow = Q + ((size_t)i*B_ + m)*C_ + h*D_;
  v8f acc0 = {}, acc1 = {};
  #pragma unroll
  for (int k0 = 0; k0 < D_; k0 += 32) {
    v16bf a, b0, b1;
    const float* ap = Arow + k0 + koffA;
    #pragma unroll
    for (int e = 0; e < 8; ++e) { a[e] = (__bf16)ap[e]; a[8+e] = (__bf16)ap[8+e]; }
    const float* bp = kw + ((size_t)i*C_ + h*D_ + k0 + koffB)*C_ + n0;  // B row-major: stride C_
    #pragma unroll
    for (int e = 0; e < 16; ++e) {
      b0[e] = (__bf16)bp[(size_t)e*C_];
      b1[e] = (__bf16)bp[(size_t)e*C_ + 16];
    }
    acc0 = __builtin_amdgcn_wmma_f32_16x16x32_bf16(false, a, false, b0, (short)0, acc0, false, false);
    acc1 = __builtin_amdgcn_wmma_f32_16x16x32_bf16(false, a, false, b1, (short)0, acc1, false, false);
  }
  #pragma unroll
  for (int rr = 0; rr < 8; ++rr) {
    int row = mt*16 + half*8 + rr;
    W[(((size_t)i*B_ + row)*H_ + h)*C_ + n0]      = acc0[rr];
    W[(((size_t)i*B_ + row)*H_ + h)*C_ + n0 + 16] = acc1[rr];
  }
}

// ---------------------------------------------------------------------------
// Kernel A3: t[i][b][h] = sum_d Q[i][b][h*192+d] * kb[i][h*192+d]
// One wave per output (4608 outputs).
// ---------------------------------------------------------------------------
__global__ __launch_bounds__(256) void k_tbias(
    const float* __restrict__ Q, const float* __restrict__ kb, float* __restrict__ T)
{
  const int tid = threadIdx.x, wid = tid >> 5, lane = tid & 31;
  const int o = blockIdx.x*8 + wid;       // ((i*128+b)*4+h)
  const int i = o >> 9;
  const int b = (o >> 2) & 127;
  const int h = o & 3;
  const float* q   = Q  + ((size_t)i*B_ + b)*C_ + h*D_;
  const float* kbp = kb + i*C_ + h*D_;
  float s = 0.f;
  #pragma unroll
  for (int k = 0; k < 6; ++k) { int d = lane + 32*k; s += q[d]*kbp[d]; }
  #pragma unroll
  for (int off = 16; off >= 1; off >>= 1) s += __shfl_xor(s, off, 32);
  if (lane == 0) T[o] = s;
}

// ---------------------------------------------------------------------------
// Kernel A4: GN[g][b][:] = normalize(G[g][b][:])  (eps 1e-12 like F.normalize)
// ---------------------------------------------------------------------------
__global__ __launch_bounds__(256) void k_gnorm(
    const float* __restrict__ g0, const float* __restrict__ g1, const float* __restrict__ g2,
    float* __restrict__ GN)
{
  __shared__ float red[256];
  const int gi = blockIdx.x / B_, b = blockIdx.x % B_;
  const float* g = (gi==0) ? g0 : (gi==1) ? g1 : g2;
  const float* row = g + (size_t)b*C_;
  const int tid = threadIdx.x;
  float s = 0.f;
  for (int c = tid; c < C_; c += 256) { float x = row[c]; s += x*x; }
  red[tid] = s; __syncthreads();
  for (int st = 128; st > 0; st >>= 1) { if (tid < st) red[tid] += red[tid+st]; __syncthreads(); }
  const float inv = 1.f / fmaxf(sqrtf(red[0]), 1e-12f);
  for (int c = tid; c < C_; c += 256) GN[((size_t)gi*B_ + b)*C_ + c] = row[c]*inv;
}

// ---------------------------------------------------------------------------
// Kernel B: streaming logits pass. One block per (modality pi, batch b).
// Per token row (768 f32): 12 attn logits, 3 cos numerators, 1 self-norm.
// w / gn vectors staged in LDS (46 KB). HBM-bound; prefetch next row
// (lowered to global_prefetch_b8) while computing the current one.
// ---------------------------------------------------------------------------
__global__ __launch_bounds__(256) void k_logits(
    const float* __restrict__ t0, const float* __restrict__ t1, const float* __restrict__ t2,
    const float* __restrict__ W, const float* __restrict__ T, const float* __restrict__ GN,
    float* __restrict__ LG, float* __restrict__ CS)
{
  __shared__ __align__(16) float s_w[12*C_];
  __shared__ __align__(16) float s_g[3*C_];
  __shared__ float s_t[12];
  const int pi = blockIdx.x / B_;
  const int b  = blockIdx.x % B_;
  const int tid = threadIdx.x;
  const float* tok = (pi==0) ? t0 : (pi==1) ? t1 : t2;
  for (int idx = tid; idx < 12*C_; idx += 256) {
    int jh = idx / C_, c = idx % C_;
    int i = pi*3 + (jh>>2), h = jh & 3;
    s_w[idx] = W[(((size_t)i*B_ + b)*H_ + h)*C_ + c];
  }
  for (int idx = tid; idx < 3*C_; idx += 256) {
    int j = idx / C_, c = idx % C_;
    s_g[idx] = GN[((size_t)c_gi[pi*3+j]*B_ + b)*C_ + c];
  }
  if (tid < 12) {
    int j = tid>>2, h = tid&3;
    s_t[tid] = T[(((size_t)(pi*3+j))*B_ + b)*H_ + h];
  }
  __syncthreads();
  const int wid = tid >> 5, lane = tid & 31;
  const float4* sw4 = reinterpret_cast<const float4*>(s_w);
  const float4* sg4 = reinterpret_cast<const float4*>(s_g);
  const float SCALE = 0.07216878364870322f;   // 192^-0.5
  for (int n = wid; n < N_; n += 8) {
    const float4* row4 = reinterpret_cast<const float4*>(tok + ((size_t)b*N_ + n)*C_);
    // prefetch this wave's next row (n+8): 32 lanes x 6 steps cover all 3 KB
    const float* nxt = tok + ((size_t)b*N_ + (n + 8 < N_ ? n + 8 : n))*C_;
    float acc[16];
    #pragma unroll
    for (int a = 0; a < 16; ++a) acc[a] = 0.f;
    #pragma unroll
    for (int k = 0; k < 6; ++k) {
      int i4 = lane + 32*k;
      __builtin_prefetch(nxt + 4*i4, 0, 0);
      float4 p = row4[i4];
      #pragma unroll
      for (int jh = 0; jh < 12; ++jh) {
        float4 wv = sw4[jh*192 + i4];
        acc[jh] += p.x*wv.x + p.y*wv.y + p.z*wv.z + p.w*wv.w;
      }
      #pragma unroll
      for (int j = 0; j < 3; ++j) {
        float4 gv = sg4[j*192 + i4];
        acc[12+j] += p.x*gv.x + p.y*gv.y + p.z*gv.z + p.w*gv.w;
      }
      acc[15] += p.x*p.x + p.y*p.y + p.z*p.z + p.w*p.w;
    }
    #pragma unroll
    for (int off = 16; off >= 1; off >>= 1) {
      #pragma unroll
      for (int a = 0; a < 16; ++a) acc[a] += __shfl_xor(acc[a], off, 32);
    }
    if (lane < 12) {
      LG[(((size_t)pi*B_ + b)*12 + lane)*N_ + n] = (acc[lane] + s_t[lane]) * SCALE;
    } else if (lane < 15) {
      float norm = fmaxf(sqrtf(acc[15]), 1e-12f);
      CS[(((size_t)pi*B_ + b)*3 + (lane-12))*N_ + n] = acc[lane] / norm;
    }
  }
}

// ---------------------------------------------------------------------------
// Kernel C: softmax over N per head, score = cos * mean_h, minmax, modal-mix,
// stable top-118 rank -> mask. One block per (pi, b).
// ---------------------------------------------------------------------------
__global__ __launch_bounds__(256) void k_select(
    const float* __restrict__ LG, const float* __restrict__ CS,
    const float* __restrict__ mw, float* __restrict__ maskOut)
{
  __shared__ float red[256];
  __shared__ float sc[3][N_];
  __shared__ float fin[N_];
  const int pi = blockIdx.x / B_, b = blockIdx.x % B_;
  const int tid = threadIdx.x;
  const bool v = tid < N_;
  for (int j = 0; j < 3; ++j) {
    float ssum = 0.f;
    for (int h = 0; h < 4; ++h) {
      float x = v ? LG[(((size_t)pi*B_ + b)*12 + j*4 + h)*N_ + tid] : -INFINITY;
      red[tid] = x; __syncthreads();
      for (int s = 128; s > 0; s >>= 1) { if (tid < s) red[tid] = fmaxf(red[tid], red[tid+s]); __syncthreads(); }
      float mx = red[0]; __syncthreads();
      float e = v ? __expf(x - mx) : 0.f;
      red[tid] = e; __syncthreads();
      for (int s = 128; s > 0; s >>= 1) { if (tid < s) red[tid] += red[tid+s]; __syncthreads(); }
      float sm = red[0]; __syncthreads();
      ssum += e / sm;
    }
    float cosv = v ? CS[(((size_t)pi*B_ + b)*3 + j)*N_ + tid] : 0.f;
    float score = cosv * ssum * 0.25f;
    red[tid] = v ? score : INFINITY; __syncthreads();
    for (int s = 128; s > 0; s >>= 1) { if (tid < s) red[tid] = fminf(red[tid], red[tid+s]); __syncthreads(); }
    float mn = red[0]; __syncthreads();
    red[tid] = v ? score : -INFINITY; __syncthreads();
    for (int s = 128; s > 0; s >>= 1) { if (tid < s) red[tid] = fmaxf(red[tid], red[tid+s]); __syncthreads(); }
    float mx = red[0]; __syncthreads();
    if (v) sc[j][tid] = (score - mn) / (mx - mn + 1e-8f);
    __syncthreads();
  }
  float w0 = mw[pi*3+0], w1 = mw[pi*3+1], w2 = mw[pi*3+2];
  float wm = fmaxf(w0, fmaxf(w1, w2));
  float e0 = __expf(w0-wm), e1 = __expf(w1-wm), e2 = __expf(w2-wm);
  float es = e0 + e1 + e2;
  if (v) fin[tid] = (e0*sc[0][tid] + e1*sc[1][tid] + e2*sc[2][tid]) / es;
  __syncthreads();
  if (v) {
    float me = fin[tid];
    int cnt = 0;
    for (int m = 0; m < N_; ++m) {
      float o = fin[m];
      cnt += (o > me) || (o == me && m < tid);   // stable top-k tie-break
    }
    maskOut[((size_t)pi*B_ + b)*N_ + tid] = (cnt < KEEP_) ? 1.f : 0.f;
  }
}

// ---------------------------------------------------------------------------
// Kernel D: out = tokens * mask, float4 stream (read 231 MB, write 231 MB)
// ---------------------------------------------------------------------------
__global__ __launch_bounds__(256) void k_apply(
    const float* __restrict__ t0, const float* __restrict__ t1, const float* __restrict__ t2,
    const float* __restrict__ maskBuf, float* __restrict__ out)
{
  const size_t PER4 = (size_t)B_*N_*C_/4;
  const size_t TOT4 = 3*PER4;
  float4* out4 = reinterpret_cast<float4*>(out);
  for (size_t idx = (size_t)blockIdx.x*blockDim.x + threadIdx.x; idx < TOT4;
       idx += (size_t)gridDim.x*blockDim.x) {
    size_t pi  = idx / PER4;
    size_t rem = idx - pi*PER4;
    size_t bn  = rem / (C_/4);
    const float4* src = (pi==0) ? reinterpret_cast<const float4*>(t0)
                      : (pi==1) ? reinterpret_cast<const float4*>(t1)
                                : reinterpret_cast<const float4*>(t2);
    float mk = maskBuf[pi*(size_t)(B_*N_) + bn];
    float4 p = src[rem];
    float4 o; o.x = p.x*mk; o.y = p.y*mk; o.z = p.z*mk; o.w = p.w*mk;
    out4[idx] = o;
  }
}

extern "C" void kernel_launch(void* const* d_in, const int* in_sizes, int n_in,
                              void* d_out, int out_size, void* d_ws, size_t ws_size,
                              hipStream_t stream) {
  const float* rgb  = (const float*)d_in[0];
  const float* ni   = (const float*)d_in[1];
  const float* ti   = (const float*)d_in[2];
  const float* grgb = (const float*)d_in[3];
  const float* gni  = (const float*)d_in[4];
  const float* gti  = (const float*)d_in[5];
  const float* qw   = (const float*)d_in[6];
  const float* qb   = (const float*)d_in[7];
  const float* kw   = (const float*)d_in[8];
  const float* kb   = (const float*)d_in[9];
  const float* mw   = (const float*)d_in[10];

  float* ws = (float*)d_ws;
  float* Q  = ws;               // 9*128*768              = 884736
  float* W  = ws + 884736;      // 9*128*4*768            = 3538944
  float* T  = ws + 4423680;     // 9*128*4                = 4608
  float* GN = ws + 4428288;     // 3*128*768              = 294912
  float* LG = ws + 4723200;     // 3*128*12*196           = 903168
  float* CS = ws + 5626368;     // 3*128*3*196            = 225792  (total ~23.4 MB)

  float* out = (float*)d_out;
  float* maskOut = out + (size_t)3*B_*N_*C_;   // m_rgb, m_nir, m_tir

  k_qproj <<<dim3(9*8*24),  dim3(32),  0, stream>>>(grgb, gni, gti, qw, qb, Q);
  k_gnorm <<<dim3(3*B_),    dim3(256), 0, stream>>>(grgb, gni, gti, GN);
  k_wproj <<<dim3(36*8*24), dim3(32),  0, stream>>>(Q, kw, W);
  k_tbias <<<dim3(576),     dim3(256), 0, stream>>>(Q, kb, T);
  k_logits<<<dim3(3*B_),    dim3(256), 0, stream>>>(rgb, ni, ti, W, T, GN, LG, CS);
  k_select<<<dim3(3*B_),    dim3(256), 0, stream>>>(LG, CS, mw, maskOut);
  k_apply <<<dim3(8192),    dim3(256), 0, stream>>>(rgb, ni, ti, maskOut, out);
}
